// BertDecoderLayerNoMemoryUntied_16140487098849
// MI455X (gfx1250) — matrix operand order repaired and checked by reading, not compiled, not run on hardware
//
#include <hip/hip_runtime.h>
#include <hip/hip_bf16.h>

typedef __bf16 bf16;
typedef __attribute__((ext_vector_type(16))) __bf16 v16bf;
typedef __attribute__((ext_vector_type(8)))  __bf16 v8bf;
typedef __attribute__((ext_vector_type(8)))  float  v8f;
typedef __attribute__((ext_vector_type(4)))  unsigned int u32x4;
typedef __attribute__((ext_vector_type(8)))  int          i32x8;
typedef __attribute__((ext_vector_type(4)))  int          i32x4;

union V16 { v16bf v; v8bf h[2]; };

#define Dmodel 1024
#define NHEAD  16
#define HDIM   64

#if __has_builtin(__builtin_amdgcn_tensor_load_to_lds) && __has_builtin(__builtin_amdgcn_s_wait_tensorcnt)
#define HAVE_TDM 1
#else
#define HAVE_TDM 0
#endif

// ---------------------------------------------------------------- fp32 -> bf16
__global__ void cvt_f32_bf16(const float* __restrict__ in, bf16* __restrict__ out, int n) {
  int i = blockIdx.x * 256 + threadIdx.x;
  if (i < n) out[i] = (bf16)in[i];
}

// ---------------------------------------------------------------- GEMM: Y = X @ W^T + bias
// X: [M,K] bf16 row-major.  W: [Dout,K] bf16 row-major (torch Linear weight).
// Block = 4 waves = one 64x64 output tile (wave w owns rows w*16..w*16+15).
// W slab (64 rows x 32 K-halves) is staged into LDS, double buffered:
//   TDM (tensor_load_to_lds + s_wait_tensorcnt) when available, else coop copy.
// mode 0: fp32 row-major out. mode 1: bf16 row-major out.
// mode 2: bf16 transposed-per-batch out: out[b*Dout*L + n*L + l] (for V).
__global__ void gemm_xwt(const bf16* __restrict__ X, const bf16* __restrict__ W,
                         const float* __restrict__ bias,
                         float* __restrict__ outF, bf16* __restrict__ outB,
                         int M, int K, int Dout, int L, int mode) {
  __shared__ bf16 wslab[2][64][32];          // 2 x 4KB double buffer
  const int tid  = threadIdx.x;              // 128
  const int lane = tid & 31;
  const int wid  = tid >> 5;
  const int nbt  = Dout / 64;
  const int mb   = blockIdx.x / nbt;
  const int nb   = blockIdx.x % nbt;
  const int half = lane >> 4;
  const int l15  = lane & 15;
  const int m    = mb * 64 + wid * 16 + l15; // A-layout row carried by this lane
  const bf16* xrow = X + (size_t)m * K;

  auto stage = [&](int ks, int bufsel) {
#if HAVE_TDM
    if (wid == 0) {
      const bf16* gtile = W + (size_t)(nb * 64) * K + (size_t)ks * 32;
      unsigned long long ga = (unsigned long long)(uintptr_t)gtile;
      unsigned int la = (unsigned int)(uintptr_t)(&wslab[bufsel][0][0]);
      u32x4 g0;
      g0[0] = 1u;                                           // count=1, user mode
      g0[1] = la;                                           // lds_addr
      g0[2] = (unsigned int)ga;                             // global_addr[31:0]
      g0[3] = (unsigned int)((ga >> 32) & 0x1FFFFFFu) | (2u << 30); // addr[56:32] | type=2
      i32x8 g1;
      g1[0] = 0x00010000;                                   // data_size=2B; no mask/pad
      g1[1] = (int)((K & 0xFFFF) << 16);                    // tensor_dim0 lo16 @bit48
      g1[2] = (int)(((K >> 16) & 0xFFFF) | ((Dout & 0xFFFF) << 16));  // dim0 hi | dim1 lo
      g1[3] = (int)(((Dout >> 16) & 0xFFFF) | (32 << 16));  // dim1 hi | tile_dim0=32
      g1[4] = 64;                                           // tile_dim1=64, tile_dim2=0
      g1[5] = (int)K;                                       // tensor_dim0_stride lo32
      g1[6] = 0;                                            // stride0 hi | stride1 lo
      g1[7] = 0;
      i32x4 gz; gz[0] = 0; gz[1] = 0; gz[2] = 0; gz[3] = 0;
      i32x8 gz8;
      gz8[0] = 0; gz8[1] = 0; gz8[2] = 0; gz8[3] = 0;
      gz8[4] = 0; gz8[5] = 0; gz8[6] = 0; gz8[7] = 0;
      __builtin_amdgcn_tensor_load_to_lds(g0, g1, gz, gz, gz8, 0);
    }
#else
    const bf16* gt = W + (size_t)(nb * 64 + (tid >> 1)) * K + ks * 32 + (tid & 1) * 16;
    *(v8bf*)(&wslab[bufsel][tid >> 1][(tid & 1) * 16]) = *(const v8bf*)gt;
#endif
  };

  v8f acc[4] = {};
  const int nSlabs = K / 32;
  stage(0, 0);
  int buf = 0;
  for (int ks = 0; ks < nSlabs; ++ks) {
    if (ks + 1 < nSlabs) stage(ks + 1, buf ^ 1);
#if HAVE_TDM
    if (wid == 0) {
      if (ks + 1 < nSlabs) __builtin_amdgcn_s_wait_tensorcnt(1);  // cur slab done (in-order)
      else                 __builtin_amdgcn_s_wait_tensorcnt(0);
    }
#endif
    __syncthreads();
    const int k0 = ks * 32;
    V16 a;   // A 16x32: lane<16 holds K {k0..+15}; lane>=16 holds K {k0+16..+31}
    a.h[0] = *(const v8bf*)(xrow + k0 + 16 * half);
    a.h[1] = *(const v8bf*)(xrow + k0 + 16 * half + 8);
    const bf16* slab = &wslab[buf][0][0];
#pragma unroll
    for (int t = 0; t < 4; ++t) {
      // B 32x16: lane c holds col (c&15) of tile t, contiguous K from 16*(c>>4)
      const bf16* wl = slab + (t * 16 + l15) * 32 + 16 * half;
      V16 b;
      b.h[0] = *(const v8bf*)(wl);
      b.h[1] = *(const v8bf*)(wl + 8);
      acc[t] = __builtin_amdgcn_wmma_f32_16x16x32_bf16(false, a.v, false, b.v,
                                                       (short)0, acc[t], false, false);
    }
    __syncthreads();
    buf ^= 1;
  }
#pragma unroll
  for (int t = 0; t < 4; ++t) {
    const int n = nb * 64 + t * 16 + l15;
    const float bv = bias ? bias[n] : 0.0f;
#pragma unroll
    for (int j = 0; j < 8; ++j) {           // C layout: row = j + 8*half, col = l15
      const int row = mb * 64 + wid * 16 + j + 8 * half;
      const float val = acc[t][j] + bv;
      if (mode == 0)      outF[(size_t)row * Dout + n] = val;
      else if (mode == 1) outB[(size_t)row * Dout + n] = (bf16)val;
      else {
        const int b = row / L, l = row % L;
        outB[((size_t)b * Dout + n) * (size_t)L + l] = (bf16)val;
      }
    }
  }
}

// ---------------------------------------------------------------- flash attention
// One wave per (batch, head, 16-row q tile); streams k in blocks of 32.
// Q,K: [N,L,D] bf16 row-major.  VT: [N,D,L] bf16 (per-batch transposed).  out: fp32 [N,Lq,D].
#define AWAVES 4
__global__ void flash_attn(const bf16* __restrict__ Q, const bf16* __restrict__ Kb,
                           const bf16* __restrict__ VT, const float* __restrict__ mask,
                           float* __restrict__ out, int Lq, int Lk, int causal) {
  __shared__ bf16 pbuf[AWAVES][16 * 32 + 16];
  const int lane = threadIdx.x & 31;
  const int wid  = threadIdx.x >> 5;
  const int wave = blockIdx.x * AWAVES + wid;
  const int qtiles = Lq / 16;
  const int b   = wave / (NHEAD * qtiles);
  const int rm_ = wave % (NHEAD * qtiles);
  const int h   = rm_ / qtiles;
  const int qt  = rm_ % qtiles;
  const int half = lane >> 4;
  const int l15  = lane & 15;

  const bf16* qrow = Q + ((size_t)b * Lq + qt * 16 + l15) * Dmodel + h * HDIM;
  V16 qa0, qa1;
  qa0.h[0] = *(const v8bf*)(qrow +      16 * half);
  qa0.h[1] = *(const v8bf*)(qrow + 8  + 16 * half);
  qa1.h[0] = *(const v8bf*)(qrow + 32 + 16 * half);
  qa1.h[1] = *(const v8bf*)(qrow + 40 + 16 * half);

  v8f acc[4] = {};
  float mrun[8], lrun[8];
#pragma unroll
  for (int j = 0; j < 8; ++j) { mrun[j] = -3.0e38f; lrun[j] = 0.0f; }

  const float sc = 0.125f;                  // 1/sqrt(64)
  const int kmax = causal ? (qt * 16 + 16) : Lk;
  for (int kb = 0; kb < kmax; kb += 32) {
    v8f s[2] = {};
#pragma unroll
    for (int nt2 = 0; nt2 < 2; ++nt2) {
      const int kpos = kb + nt2 * 16 + l15;
      const bf16* krow = Kb + ((size_t)b * Lk + kpos) * Dmodel + h * HDIM + 16 * half;
      V16 b0, b1;
      b0.h[0] = *(const v8bf*)(krow);       b0.h[1] = *(const v8bf*)(krow + 8);
      b1.h[0] = *(const v8bf*)(krow + 32);  b1.h[1] = *(const v8bf*)(krow + 40);
      s[nt2] = __builtin_amdgcn_wmma_f32_16x16x32_bf16(false, qa0.v, false, b0.v,
                                                       (short)0, s[nt2], false, false);
      s[nt2] = __builtin_amdgcn_wmma_f32_16x16x32_bf16(false, qa1.v, false, b1.v,
                                                       (short)0, s[nt2], false, false);
    }
#pragma unroll
    for (int nt2 = 0; nt2 < 2; ++nt2) {
      const int col = kb + nt2 * 16 + l15;
      const float mv = mask[(size_t)b * Lk + col];
#pragma unroll
      for (int j = 0; j < 8; ++j) {
        const int row = qt * 16 + j + 8 * half;
        float mm = mv;
        if (causal && col > row) mm = 0.0f;
        s[nt2][j] = s[nt2][j] * sc + (1.0f - mm) * (-10000.0f);
      }
    }
    float corr[8];
#pragma unroll
    for (int j = 0; j < 8; ++j) {
      float v = fmaxf(s[0][j], s[1][j]);
      v = fmaxf(v, __shfl_xor(v, 1)); v = fmaxf(v, __shfl_xor(v, 2));
      v = fmaxf(v, __shfl_xor(v, 4)); v = fmaxf(v, __shfl_xor(v, 8));
      const float mn = fmaxf(mrun[j], v);
      corr[j] = __expf(mrun[j] - mn);
      mrun[j] = mn;
    }
#pragma unroll
    for (int j = 0; j < 8; ++j) {
      s[0][j] = __expf(s[0][j] - mrun[j]);
      s[1][j] = __expf(s[1][j] - mrun[j]);
      float v = s[0][j] + s[1][j];
      v += __shfl_xor(v, 1); v += __shfl_xor(v, 2);
      v += __shfl_xor(v, 4); v += __shfl_xor(v, 8);
      lrun[j] = lrun[j] * corr[j] + v;
#pragma unroll
      for (int t = 0; t < 4; ++t) acc[t][j] *= corr[j];
    }
    bf16* pw = pbuf[wid];
#pragma unroll
    for (int nt2 = 0; nt2 < 2; ++nt2) {
#pragma unroll
      for (int j = 0; j < 8; ++j)
        pw[(j + 8 * half) * 32 + nt2 * 16 + l15] = (bf16)s[nt2][j];
    }
    asm volatile("s_wait_dscnt 0" ::: "memory");
    V16 pa;
    pa.h[0] = *(const v8bf*)(pw + l15 * 32 +     16 * half);
    pa.h[1] = *(const v8bf*)(pw + l15 * 32 + 8 + 16 * half);
#pragma unroll
    for (int t = 0; t < 4; ++t) {
      const int n = t * 16 + l15;
      const bf16* vrow = VT + ((size_t)b * Dmodel + h * HDIM + n) * (size_t)Lk + kb + 16 * half;
      V16 vb;
      vb.h[0] = *(const v8bf*)(vrow);
      vb.h[1] = *(const v8bf*)(vrow + 8);
      acc[t] = __builtin_amdgcn_wmma_f32_16x16x32_bf16(false, pa.v, false, vb.v,
                                                       (short)0, acc[t], false, false);
    }
  }
#pragma unroll
  for (int t = 0; t < 4; ++t) {
    const int n = h * HDIM + t * 16 + l15;
#pragma unroll
    for (int j = 0; j < 8; ++j) {
      const int row = qt * 16 + j + 8 * half;
      out[((size_t)b * Lq + row) * Dmodel + n] = acc[t][j] / lrun[j];
    }
  }
}

// ---------------------------------------------------------------- y = LN(x1+x2)*w + b
__global__ void add_ln(const float* __restrict__ x1, const float* __restrict__ x2,
                       const float* __restrict__ w, const float* __restrict__ bias,
                       float* __restrict__ outF, bf16* __restrict__ outB, int D) {
  __shared__ float red[8];
  const int row = blockIdx.x;
  const int tid = threadIdx.x;                 // 256 threads = 8 waves
  const float inv = 1.0f / (float)D;
  const float* a = x1 + (size_t)row * D;
  const float* c = x2 + (size_t)row * D;
  float s = 0.0f;
  for (int i = tid; i < D; i += 256) s += a[i] + c[i];
  for (int o = 16; o >= 1; o >>= 1) s += __shfl_xor(s, o);
  if ((tid & 31) == 0) red[tid >> 5] = s;
  __syncthreads();
  float tot = 0.0f;
  for (int i = 0; i < 8; ++i) tot += red[i];
  const float mean = tot * inv;
  __syncthreads();
  float v = 0.0f;
  for (int i = tid; i < D; i += 256) { float d = a[i] + c[i] - mean; v += d * d; }
  for (int o = 16; o >= 1; o >>= 1) v += __shfl_xor(v, o);
  if ((tid & 31) == 0) red[tid >> 5] = v;
  __syncthreads();
  float vt = 0.0f;
  for (int i = 0; i < 8; ++i) vt += red[i];
  const float rstd = rsqrtf(vt * inv + 1e-12f);
  for (int i = tid; i < D; i += 256) {
    const float y = w[i] * (a[i] + c[i] - mean) * rstd + bias[i];
    if (outF) outF[(size_t)row * D + i] = y;
    if (outB) outB[(size_t)row * D + i] = (bf16)y;
  }
}

// ----------------------------------------------------------------
extern "C" void kernel_launch(void* const* d_in, const int* in_sizes, int n_in,
                              void* d_out, int out_size, void* d_ws, size_t ws_size,
                              hipStream_t stream) {
  (void)in_sizes; (void)n_in; (void)out_size; (void)ws_size;
  const int N = 4, LT = 1024, LV = 1024, D = Dmodel;
  const int M = N * LT;

  const float* dec   = (const float*)d_in[0];
  const float* enc   = (const float*)d_in[1];
  const float* dmask = (const float*)d_in[2];
  const float* emask = (const float*)d_in[3];
  const float* wP[7]; for (int i = 0; i < 7; ++i) wP[i] = (const float*)d_in[4 + i];
  const float* sa_qb = (const float*)d_in[11];
  const float* sa_kb = (const float*)d_in[12];
  const float* sa_vb = (const float*)d_in[13];
  const float* ca_qb = (const float*)d_in[14];
  const float* ca_kb = (const float*)d_in[15];
  const float* ca_vb = (const float*)d_in[16];
  const float* out_b = (const float*)d_in[17];
  const float* n1_b  = (const float*)d_in[18];
  const float* n2_b  = (const float*)d_in[19];
  const float* oln_b = (const float*)d_in[20];
  const float* n1_w  = (const float*)d_in[21];
  const float* n2_w  = (const float*)d_in[22];
  const float* oln_w = (const float*)d_in[23];

  char* ws = (char*)d_ws;
  size_t off = 0;
  auto alloc = [&](size_t bytes) -> void* {
    off = (off + 255) & ~(size_t)255;
    void* p = ws + off;
    off += bytes;
    return p;
  };
  const size_t EB = (size_t)N * LT * D;
  bf16* decB = (bf16*)alloc(EB * 2);
  bf16* encB = (bf16*)alloc(EB * 2);
  bf16* wB[7]; for (int i = 0; i < 7; ++i) wB[i] = (bf16*)alloc((size_t)D * D * 2);
  bf16* qB   = (bf16*)alloc(EB * 2);
  bf16* kB   = (bf16*)alloc(EB * 2);
  bf16* vTB  = (bf16*)alloc(EB * 2);
  bf16* xB   = (bf16*)alloc(EB * 2);
  float* ctxF = (float*)alloc(EB * 4);
  float* aF   = (float*)alloc(EB * 4);
  float* cF   = (float*)alloc(EB * 4);

  const int nel = (int)EB;
  cvt_f32_bf16<<<(nel + 255) / 256, 256, 0, stream>>>(dec, decB, nel);
  cvt_f32_bf16<<<(nel + 255) / 256, 256, 0, stream>>>(enc, encB, nel);
  for (int i = 0; i < 7; ++i)
    cvt_f32_bf16<<<(D * D + 255) / 256, 256, 0, stream>>>(wP[i], wB[i], D * D);

  const int gemmBlocks = (M / 64) * (D / 64);          // 64x64 tile per 4-wave block
  const int attnBlocks = (N * NHEAD * (LT / 16)) / AWAVES;

  // ---- self attention
  gemm_xwt<<<gemmBlocks, 128, 0, stream>>>(decB, wB[0], sa_qb, nullptr, qB,  M, D, D, LT, 1);
  gemm_xwt<<<gemmBlocks, 128, 0, stream>>>(decB, wB[1], sa_kb, nullptr, kB,  M, D, D, LT, 1);
  gemm_xwt<<<gemmBlocks, 128, 0, stream>>>(decB, wB[2], sa_vb, nullptr, vTB, M, D, D, LT, 2);
  flash_attn<<<attnBlocks, 32 * AWAVES, 0, stream>>>(qB, kB, vTB, dmask, ctxF, LT, LT, 1);
  add_ln<<<M, 256, 0, stream>>>(ctxF, dec, n1_w, n1_b, aF, xB, D);

  // ---- cross attention
  gemm_xwt<<<gemmBlocks, 128, 0, stream>>>(xB,   wB[3], ca_qb, nullptr, qB,  M, D, D, LT, 1);
  gemm_xwt<<<gemmBlocks, 128, 0, stream>>>(encB, wB[4], ca_kb, nullptr, kB,  M, D, D, LV, 1);
  gemm_xwt<<<gemmBlocks, 128, 0, stream>>>(encB, wB[5], ca_vb, nullptr, vTB, M, D, D, LV, 2);
  flash_attn<<<attnBlocks, 32 * AWAVES, 0, stream>>>(qB, kB, vTB, emask, ctxF, LT, LV, 0);
  add_ln<<<M, 256, 0, stream>>>(ctxF, aF, n2_w, n2_b, cF, xB, D);

  // ---- output dense + final LN
  gemm_xwt<<<gemmBlocks, 128, 0, stream>>>(xB, wB[6], out_b, ctxF, nullptr, M, D, D, LT, 0);
  add_ln<<<M, 256, 0, stream>>>(ctxF, cF, oln_w, oln_b, (float*)d_out, nullptr, D);
}
